// CrossAttention_31078383354530
// MI455X (gfx1250) — matrix operand order repaired and checked by reading, not compiled, not run on hardware
//
#include <hip/hip_runtime.h>
#include <hip/hip_bf16.h>

// ---------------------------------------------------------------------------
// CDNA5 (gfx1250) cross-attention, all matmuls via v_wmma_f32_16x16x32_bf16.
// Mean-centered tau logits collapse to a plain scale (softmax shift-invariance):
//   scale = 1.5 / sqrt(128)
// Row-major tile staging uses GLOBAL_LOAD_ASYNC_TO_LDS_B128 (ASYNCcnt path,
// bypasses VGPRs); transposed stagings keep the VGPR scatter path.
// ---------------------------------------------------------------------------

typedef __attribute__((ext_vector_type(16))) __bf16 v16bf;
typedef __attribute__((ext_vector_type(8)))  float  v8f;

union Bf8 { uint4 v; __bf16 h[8]; };

__device__ __forceinline__ v8f wmma_bf16(const v16bf& a, const v16bf& b, const v8f& c) {
  return __builtin_amdgcn_wmma_f32_16x16x32_bf16(false, a, false, b, (short)0, c, false, false);
}

// Async global -> LDS copy of 16 bytes per lane (CDNA5 VGLOBAL async op).
__device__ __forceinline__ void async_copy_b128(const void* gptr, const void* lptr) {
  uint32_t lds = (uint32_t)(uintptr_t)lptr;  // generic->LDS: low 32 bits
  asm volatile("global_load_async_to_lds_b128 %0, %1, off"
               :: "v"(lds), "v"(gptr) : "memory");
}
__device__ __forceinline__ void wait_asynccnt0() {
  asm volatile("s_wait_asynccnt 0x0" ::: "memory");
}

// ------------------------------ fp32 -> bf16 -------------------------------
__global__ void f32_to_bf16_kernel(const float* __restrict__ in,
                                   __bf16* __restrict__ out, int n) {
  int i = blockIdx.x * blockDim.x + threadIdx.x;
  if (i < n) out[i] = (__bf16)in[i];
}

// ------------------------------- WMMA GEMM ---------------------------------
// C(M,N) = A(M,K) @ B(K,N), A/B bf16 row-major.
// EPI==0: bf16 head-split store  Cb[(h*M + row)*128 + (col&127)], h = col>>7
// EPI==1: f32 store with bias    Cf[row*N + col] = acc + bias[col]
template <int EPI>
__global__ void __launch_bounds__(256)
gemm_wmma(const __bf16* __restrict__ A, const __bf16* __restrict__ B,
          __bf16* __restrict__ Cb, float* __restrict__ Cf,
          const float* __restrict__ bias, int M, int N, int K) {
  constexpr int BM = 128, BN = 128, BK = 32;
  __shared__ __bf16 sA[BM * BK];   // [row][k]   row-major
  __shared__ __bf16 sBt[BN * BK];  // [col][k]   k-contiguous (transposed)

  const int tid = threadIdx.x;
  const int lane = tid & 31;
  const int w = tid >> 5;
  const int wm = w & 3;   // 4 waves along M -> 32 rows each
  const int wn = w >> 2;  // 2 waves along N -> 64 cols each
  const int rowBase = blockIdx.y * BM;
  const int colBase = blockIdx.x * BN;
  const int r = lane & 15, hi = lane >> 4;

  v8f acc[2][4];
#pragma unroll
  for (int mi = 0; mi < 2; mi++)
#pragma unroll
    for (int ni = 0; ni < 4; ni++)
#pragma unroll
      for (int e = 0; e < 8; e++) acc[mi][ni][e] = 0.0f;

  for (int k0 = 0; k0 < K; k0 += BK) {
    __syncthreads();
    {  // A tile: 128x32, 16 bf16 per thread via async global->LDS
      int ar = tid >> 1, ac = (tid & 1) * 16;
      const __bf16* gp = A + (size_t)(rowBase + ar) * K + k0 + ac;
      async_copy_b128(gp,     &sA[ar * BK + ac]);
      async_copy_b128(gp + 8, &sA[ar * BK + ac + 8]);
    }
    {  // B tile: 32x128 -> store k-contiguous per column (needs transpose)
      int br = tid >> 3, bc = (tid & 7) * 16;
      const __bf16* gp = B + (size_t)(k0 + br) * N + colBase + bc;
      Bf8 v0, v1;
      v0.v = *(const uint4*)gp;
      v1.v = *(const uint4*)(gp + 8);
#pragma unroll
      for (int j = 0; j < 8; j++) {
        sBt[(bc + j) * BK + br]     = v0.h[j];
        sBt[(bc + 8 + j) * BK + br] = v1.h[j];
      }
    }
    wait_asynccnt0();
    __syncthreads();

    // A fragments (16x32): lane r = row, hi-half holds K=16..31
    v16bf af[2];
#pragma unroll
    for (int mi = 0; mi < 2; mi++) {
      const __bf16* p = &sA[(wm * 32 + mi * 16 + r) * BK + hi * 16];
      Bf8 t0, t1;
      t0.v = *(const uint4*)p;
      t1.v = *(const uint4*)(p + 8);
#pragma unroll
      for (int i = 0; i < 8; i++) { af[mi][i] = t0.h[i]; af[mi][8 + i] = t1.h[i]; }
    }
    // B fragments (32x16): lane r = col, hi-half holds K=16..31
    v16bf bfrag[4];
#pragma unroll
    for (int ni = 0; ni < 4; ni++) {
      const __bf16* p = &sBt[(wn * 64 + ni * 16 + r) * BK + hi * 16];
      Bf8 t0, t1;
      t0.v = *(const uint4*)p;
      t1.v = *(const uint4*)(p + 8);
#pragma unroll
      for (int i = 0; i < 8; i++) { bfrag[ni][i] = t0.h[i]; bfrag[ni][8 + i] = t1.h[i]; }
    }
#pragma unroll
    for (int mi = 0; mi < 2; mi++)
#pragma unroll
      for (int ni = 0; ni < 4; ni++)
        acc[mi][ni] = wmma_bf16(af[mi], bfrag[ni], acc[mi][ni]);
  }

  // Epilogue. C layout: VGPR e, lane -> row = e + 8*hi, col = r.
#pragma unroll
  for (int mi = 0; mi < 2; mi++)
#pragma unroll
    for (int ni = 0; ni < 4; ni++)
#pragma unroll
      for (int e = 0; e < 8; e++) {
        int gr = rowBase + wm * 32 + mi * 16 + e + hi * 8;
        int gc = colBase + wn * 64 + ni * 16 + r;
        float v = acc[mi][ni][e];
        if (EPI == 0) {
          int h = gc >> 7, d = gc & 127;
          Cb[((size_t)h * M + gr) * 128 + d] = (__bf16)v;
        } else {
          Cf[(size_t)gr * N + gc] = v + bias[gc];
        }
      }
}

// ------------------------- flash attention (per head) ----------------------
// Qh/Kh/Vh: (8 heads, 4096, 128) bf16.  Obuf: (4096, 1024) bf16, col = h*128+d.
// Grid: (32 query-blocks of 128, 8 heads), 256 threads (8 waves).
// Each wave owns 16 query rows x full d=128; block streams 32-key KV tiles.
__global__ void __launch_bounds__(256)
attn_kernel(const __bf16* __restrict__ Qh, const __bf16* __restrict__ Kh,
            const __bf16* __restrict__ Vh, __bf16* __restrict__ Obuf) {
  constexpr int NQ = 4096, DH = 128, BJ = 32;
  __shared__ __bf16 sK[BJ * DH];      // [j][d]    row-major
  __shared__ __bf16 sVt[DH * BJ];     // [d][j]    j(=K dim)-contiguous
  __shared__ __bf16 sP[8][16 * BJ];   // per-wave P bounce (C-layout -> A-layout)

  const int tid = threadIdx.x, lane = tid & 31, w = tid >> 5;
  const int r = lane & 15, hi = lane >> 4;
  const int head = blockIdx.y;
  const int qbase = blockIdx.x * 128;
  const float sc = 1.5f * 0.08838834764831845f;  // tau / sqrt(dim_head)

  const size_t hoff = (size_t)head * NQ * DH;
  const __bf16* Q = Qh + hoff;
  const __bf16* Kp = Kh + hoff;
  const __bf16* Vp = Vh + hoff;

  // Q A-fragments for this wave's 16 rows, K-dim split into 4 chunks of 32.
  v16bf qa[4];
  {
    const __bf16* qrow = Q + (size_t)(qbase + w * 16 + r) * DH;
#pragma unroll
    for (int t = 0; t < 4; t++) {
      Bf8 t0, t1;
      t0.v = *(const uint4*)(qrow + t * 32 + hi * 16);
      t1.v = *(const uint4*)(qrow + t * 32 + hi * 16 + 8);
#pragma unroll
      for (int i = 0; i < 8; i++) { qa[t][i] = t0.h[i]; qa[t][8 + i] = t1.h[i]; }
    }
  }

  float mrow[8], lrow[8];
  v8f oacc[8];
#pragma unroll
  for (int e = 0; e < 8; e++) { mrow[e] = -1e30f; lrow[e] = 0.0f; }
#pragma unroll
  for (int dt = 0; dt < 8; dt++)
#pragma unroll
    for (int e = 0; e < 8; e++) oacc[dt][e] = 0.0f;

  for (int j0 = 0; j0 < NQ; j0 += BJ) {
    __syncthreads();  // WAR on sK/sVt from previous iteration
    {  // stage K (async, row-major) and V (VGPR scatter, transposed): 32x128
      int rk = tid >> 3, c = (tid & 7) * 16;
      const __bf16* gk = Kp + (size_t)(j0 + rk) * DH + c;
      async_copy_b128(gk,     &sK[rk * DH + c]);
      async_copy_b128(gk + 8, &sK[rk * DH + c + 8]);
      const __bf16* gv = Vp + (size_t)(j0 + rk) * DH + c;
      Bf8 v0, v1;
      v0.v = *(const uint4*)gv;
      v1.v = *(const uint4*)(gv + 8);
#pragma unroll
      for (int i = 0; i < 8; i++) {
        sVt[(c + i) * BJ + rk]     = v0.h[i];
        sVt[(c + 8 + i) * BJ + rk] = v1.h[i];
      }
    }
    wait_asynccnt0();
    __syncthreads();

    // S = Q K^T : 16 x 32 (two 16x16 tiles), B[dk][j] = K[j][dk] from sK rows.
    v8f s0, s1;
#pragma unroll
    for (int e = 0; e < 8; e++) { s0[e] = 0.0f; s1[e] = 0.0f; }
#pragma unroll
    for (int t = 0; t < 4; t++) {
      v16bf kb0, kb1;
      const __bf16* p0 = &sK[(r)      * DH + t * 32 + hi * 16];
      const __bf16* p1 = &sK[(16 + r) * DH + t * 32 + hi * 16];
      Bf8 a0, a1, b0, b1;
      a0.v = *(const uint4*)p0; a1.v = *(const uint4*)(p0 + 8);
      b0.v = *(const uint4*)p1; b1.v = *(const uint4*)(p1 + 8);
#pragma unroll
      for (int i = 0; i < 8; i++) {
        kb0[i] = a0.h[i]; kb0[8 + i] = a1.h[i];
        kb1[i] = b0.h[i]; kb1[8 + i] = b1.h[i];
      }
      s0 = wmma_bf16(qa[t], kb0, s0);
      s1 = wmma_bf16(qa[t], kb1, s1);
    }

    // Online softmax. Row (e + 8*hi) lives in VGPR e across a 16-lane half.
    float rmax[8], alpha[8], rs[8];
#pragma unroll
    for (int e = 0; e < 8; e++) {
      s0[e] *= sc; s1[e] *= sc;
      rmax[e] = fmaxf(s0[e], s1[e]);
    }
#pragma unroll
    for (int off = 1; off < 16; off <<= 1)
#pragma unroll
      for (int e = 0; e < 8; e++)
        rmax[e] = fmaxf(rmax[e], __shfl_xor(rmax[e], off, 32));
#pragma unroll
    for (int e = 0; e < 8; e++) {
      float mn = fmaxf(mrow[e], rmax[e]);
      alpha[e] = __expf(mrow[e] - mn);
      mrow[e] = mn;
      s0[e] = __expf(s0[e] - mn);
      s1[e] = __expf(s1[e] - mn);
      rs[e] = s0[e] + s1[e];
    }
#pragma unroll
    for (int off = 1; off < 16; off <<= 1)
#pragma unroll
      for (int e = 0; e < 8; e++) rs[e] += __shfl_xor(rs[e], off, 32);
#pragma unroll
    for (int e = 0; e < 8; e++) lrow[e] = lrow[e] * alpha[e] + rs[e];
#pragma unroll
    for (int dt = 0; dt < 8; dt++)
#pragma unroll
      for (int e = 0; e < 8; e++) oacc[dt][e] *= alpha[e];

    // P: C-layout -> LDS (row-major 16x32) -> reload as A-fragment.
    {
      __bf16* pp = &sP[w][0];
#pragma unroll
      for (int e = 0; e < 8; e++) {
        int mr = e + hi * 8;
        pp[mr * BJ + r]      = (__bf16)s0[e];
        pp[mr * BJ + 16 + r] = (__bf16)s1[e];
      }
    }
    __syncthreads();
    v16bf pa;
    {
      const __bf16* p = &sP[w][r * BJ + hi * 16];
      Bf8 t0, t1;
      t0.v = *(const uint4*)p;
      t1.v = *(const uint4*)(p + 8);
#pragma unroll
      for (int i = 0; i < 8; i++) { pa[i] = t0.h[i]; pa[8 + i] = t1.h[i]; }
    }

    // O += P @ V, 8 d-tiles of 16 cols; B[k][d] fragments contiguous in sVt.
#pragma unroll
    for (int dt = 0; dt < 8; dt++) {
      v16bf vb;
      const __bf16* p = &sVt[(dt * 16 + r) * BJ + hi * 16];
      Bf8 t0, t1;
      t0.v = *(const uint4*)p;
      t1.v = *(const uint4*)(p + 8);
#pragma unroll
      for (int i = 0; i < 8; i++) { vb[i] = t0.h[i]; vb[8 + i] = t1.h[i]; }
      oacc[dt] = wmma_bf16(pa, vb, oacc[dt]);
    }
  }

  // Normalize and store (b, n, h*d) bf16 row-major for the output GEMM.
  float inv[8];
#pragma unroll
  for (int e = 0; e < 8; e++) inv[e] = 1.0f / lrow[e];
#pragma unroll
  for (int dt = 0; dt < 8; dt++)
#pragma unroll
    for (int e = 0; e < 8; e++) {
      int grow = qbase + w * 16 + e + hi * 8;
      int gcol = head * 128 + dt * 16 + r;
      Obuf[(size_t)grow * 1024 + gcol] = (__bf16)(oacc[dt][e] * inv[e]);
    }
}

// ------------------------------- launcher ----------------------------------
extern "C" void kernel_launch(void* const* d_in, const int* in_sizes, int n_in,
                              void* d_out, int out_size, void* d_ws, size_t ws_size,
                              hipStream_t stream) {
  (void)in_sizes; (void)n_in; (void)out_size; (void)ws_size;
  constexpr int NQ = 4096;      // b*n = b*m
  constexpr int QD = 1024, CD = 768, INNER = 1024;

  const float* x    = (const float*)d_in[0];
  const float* ctx  = (const float*)d_in[1];
  const float* Wq   = (const float*)d_in[2];
  const float* Wk   = (const float*)d_in[3];
  const float* Wv   = (const float*)d_in[4];
  const float* Wout = (const float*)d_in[5];
  const float* bout = (const float*)d_in[6];
  float* out = (float*)d_out;

  __bf16* ws = (__bf16*)d_ws;
  size_t off = 0;
  auto alloc = [&](size_t n) { __bf16* p = ws + off; off += n; return p; };
  __bf16* xb    = alloc((size_t)NQ * QD);
  __bf16* ctxb  = alloc((size_t)NQ * CD);
  __bf16* wqb   = alloc((size_t)QD * INNER);
  __bf16* wkb   = alloc((size_t)CD * INNER);
  __bf16* wvb   = alloc((size_t)CD * INNER);
  __bf16* woutb = alloc((size_t)INNER * QD);
  __bf16* Qh    = alloc((size_t)NQ * INNER);
  __bf16* Kh    = alloc((size_t)NQ * INNER);
  __bf16* Vh    = alloc((size_t)NQ * INNER);
  __bf16* Obuf  = alloc((size_t)NQ * INNER);

  auto cvt = [&](const float* in, __bf16* o, int n) {
    f32_to_bf16_kernel<<<(n + 255) / 256, 256, 0, stream>>>(in, o, n);
  };
  cvt(x,    xb,    NQ * QD);
  cvt(ctx,  ctxb,  NQ * CD);
  cvt(Wq,   wqb,   QD * INNER);
  cvt(Wk,   wkb,   CD * INNER);
  cvt(Wv,   wvb,   CD * INNER);
  cvt(Wout, woutb, INNER * QD);

  dim3 gg(INNER / 128, NQ / 128);  // (8, 32)
  gemm_wmma<0><<<gg, 256, 0, stream>>>(xb,   wqb, Qh, nullptr, nullptr, NQ, INNER, QD);
  gemm_wmma<0><<<gg, 256, 0, stream>>>(ctxb, wkb, Kh, nullptr, nullptr, NQ, INNER, CD);
  gemm_wmma<0><<<gg, 256, 0, stream>>>(ctxb, wvb, Vh, nullptr, nullptr, NQ, INNER, CD);

  attn_kernel<<<dim3(NQ / 128, 8), 256, 0, stream>>>(Qh, Kh, Vh, Obuf);

  gemm_wmma<1><<<gg, 256, 0, stream>>>(Obuf, woutb, nullptr, out, bout, NQ, QD, INNER);
}
